// Net_Obser_9354438771137
// MI455X (gfx1250) — matrix operand order repaired and checked by reading, not compiled, not run on hardware
//
#include <hip/hip_runtime.h>
#include <hip/hip_bf16.h>

typedef __attribute__((ext_vector_type(16))) _Float16 v16h;
typedef __attribute__((ext_vector_type(8)))  _Float16 v8h;
typedef __attribute__((ext_vector_type(4)))  _Float16 v4h;
typedef __attribute__((ext_vector_type(8)))  float    v8f;

#define NBATCH 256
#define PPIX   1248      // 26*48
#define CDFN   500
#define SCALE_C 233.898239415353f

// ---------------------------------------------------------------------------
// activation: 0=none 1=relu 2=selu 3=sigmoid
// ---------------------------------------------------------------------------
__device__ __forceinline__ float act_f(float x, int act) {
  if (act == 1) return x > 0.f ? x : 0.f;
  if (act == 2) {
    const float a = 1.6732632423543772f, s = 1.0507009873554805f;
    return x > 0.f ? s * x : s * a * (expf(x) - 1.f);
  }
  if (act == 3) return 1.f / (1.f + expf(-x));
  return x;
}

__device__ __forceinline__ float selu_f(float x) {
  const float a = 1.6732632423543772f, s = 1.0507009873554805f;
  return x > 0.f ? s * x : s * a * (expf(x) - 1.f);
}

// ---------------------------------------------------------------------------
// WMMA GEMM:  C[M][N] = act( A[M][K] * B[K][N] + bias[M] )
//   A strided (sAm, sAk, aBase) -> conv weights & deconv weight slices.
//   CONV=1: B is implicit-im2col gather over [Ci][NBATCH][Hin][Win].
//   AVEC=1: A rows 16B-aligned & contiguous in K -> float4 staging loads.
// 256 threads = 8 wave32 waves; block tile 128x128, K-step 32.
// Double-buffered LDS; per wave 64x32 subtile = 4x2 WMMA accumulators.
// ---------------------------------------------------------------------------
struct GemmArgs {
  const float* A; const float* B; const float* bias; float* C;
  int M, K, N;
  int sAm, sAk, aBase;
  int act;
  int Hin, Win, kh, kw, Ho, Wo;
  int HWin, NbHWin, HoWo;
};

template <int CONV, int AVEC>
__global__ __launch_bounds__(256)
void gemm_wmma_kernel(GemmArgs g) {
  __shared__ alignas(16) _Float16 As[2][128][32];
  __shared__ alignas(16) _Float16 Bs[2][128][32];
  const int tid  = threadIdx.x;
  const int lane = tid & 31;
  const int wave = tid >> 5;
  const int wm = wave >> 2;         // 0..1  (M direction)
  const int wn = wave & 3;          // 0..3  (N direction)
  const int lh = lane >> 4;
  const int lr = lane & 15;
  const int n0 = blockIdx.x * 128;
  const int m0 = blockIdx.y * 128;

  // A staging map: pass p: row (tid>>3)+p*32, 4 consecutive k at (tid&7)*4
  const int a_m  = tid >> 3;
  const int a_k4 = (tid & 7) * 4;
  // B staging map: pass p: row k=(tid>>5)+p*8, 4 consecutive n at (tid&31)*4
  const int b_k  = tid >> 5;
  const int b_n4 = (tid & 31) * 4;

  // ---- per-thread column decode, hoisted out of the K loop ----
  long  colbase[4];
  bool  colok[4];
  for (int c = 0; c < 4; ++c) {
    int ng = n0 + b_n4 + c;
    bool ok = ng < g.N;
    colok[c] = ok;
    int ngc = ok ? ng : 0;
    if (CONV) {
      int nb = ngc / g.HoWo, hw = ngc - nb * g.HoWo;
      int ho = hw / g.Wo,    wo = hw - ho * g.Wo;
      colbase[c] = (long)nb * g.HWin + (long)ho * g.Win + wo;
    } else {
      colbase[c] = ngc;
    }
  }
  // A row pointers (clamped, unconditional-load safe)
  const float* arow[4];
  bool rowok[4];
  for (int p = 0; p < 4; ++p) {
    int mg = m0 + a_m + p * 32;
    rowok[p] = mg < g.M;
    int mr = rowok[p] ? mg : (g.M - 1);
    arow[p] = g.A + (size_t)g.aBase + (size_t)mr * g.sAm;
  }
  const bool bColVec = (n0 + b_n4 + 3 < g.N);   // plain-B float4 path ok

  float ra[4][4], rb[4][4];

  auto load_tile = [&](int kb) {
    // ---- A -> registers ----
    for (int p = 0; p < 4; ++p) {
      if (AVEC && (kb + a_k4 + 3 < g.K)) {
        float4 v = *(const float4*)(arow[p] + (kb + a_k4));
        ra[p][0] = rowok[p] ? v.x : 0.f;
        ra[p][1] = rowok[p] ? v.y : 0.f;
        ra[p][2] = rowok[p] ? v.z : 0.f;
        ra[p][3] = rowok[p] ? v.w : 0.f;
      } else {
        for (int c = 0; c < 4; ++c) {
          int kg = kb + a_k4 + c;
          int kr = kg < g.K ? kg : (g.K - 1);
          float v = arow[p][(size_t)kr * g.sAk];
          ra[p][c] = (rowok[p] && kg < g.K) ? v : 0.f;
        }
      }
    }
    // ---- B -> registers ----
    for (int p = 0; p < 4; ++p) {
      int kg = kb + b_k + p * 8;
      bool kok = kg < g.K;
      int kr = kok ? kg : (g.K - 1);
      if (!CONV) {
        const float* bp = g.B + (size_t)kr * g.N;
        if (bColVec) {
          float4 v = *(const float4*)(bp + n0 + b_n4);
          rb[p][0] = kok ? v.x : 0.f;
          rb[p][1] = kok ? v.y : 0.f;
          rb[p][2] = kok ? v.z : 0.f;
          rb[p][3] = kok ? v.w : 0.f;
        } else {
          for (int c = 0; c < 4; ++c) {
            float v = bp[colbase[c]];
            rb[p][c] = (kok && colok[c]) ? v : 0.f;
          }
        }
      } else {
        int khw = g.kh * g.kw;
        int ci = kr / khw, r = kr - ci * khw;
        int i  = r / g.kw, j = r - i * g.kw;
        long offk = (long)ci * g.NbHWin + (long)i * g.Win + j;
        for (int c = 0; c < 4; ++c) {
          float v = g.B[offk + colbase[c]];
          rb[p][c] = (kok && colok[c]) ? v : 0.f;
        }
      }
    }
  };

  auto store_tile = [&](int buf) {
    for (int p = 0; p < 4; ++p) {
      v4h h;
      h[0] = (_Float16)ra[p][0]; h[1] = (_Float16)ra[p][1];
      h[2] = (_Float16)ra[p][2]; h[3] = (_Float16)ra[p][3];
      *(v4h*)&As[buf][a_m + p * 32][a_k4] = h;     // one ds_store_b64
      int k = b_k + p * 8;
      for (int c = 0; c < 4; ++c)
        Bs[buf][b_n4 + c][k] = (_Float16)rb[p][c]; // transposed [n][k]
    }
  };

  v8f acc[4][2] = {};

  load_tile(0);
  store_tile(0);
  int cur = 0;

  for (int kb = 0; kb < g.K; kb += 32) {
    const bool hasNext = (kb + 32) < g.K;
    if (hasNext) load_tile(kb + 32);     // global loads overlap WMMA below
    __syncthreads();                      // buffer `cur` fully written

    union V16 { v16h v; v8h h[2]; };
    v16h af[4], bf[2];
    for (int mf = 0; mf < 4; ++mf) {
      const _Float16* rp = &As[cur][wm * 64 + mf * 16 + lr][0];
      V16 u;
      u.h[0] = *(const v8h*)(rp + lh * 8);        // K = h*8 .. h*8+7
      u.h[1] = *(const v8h*)(rp + 16 + lh * 8);   // K = 16+h*8 ..
      af[mf] = u.v;
    }
    for (int nf = 0; nf < 2; ++nf) {
      const _Float16* rp = &Bs[cur][wn * 32 + nf * 16 + lr][0];
      V16 u;
      u.h[0] = *(const v8h*)(rp + lh * 8);        // K = h*8 .. h*8+7
      u.h[1] = *(const v8h*)(rp + 16 + lh * 8);   // K = 16+h*8 ..
      bf[nf] = u.v;
    }
    for (int mf = 0; mf < 4; ++mf)
      for (int nf = 0; nf < 2; ++nf)
        acc[mf][nf] = __builtin_amdgcn_wmma_f32_16x16x32_f16(
            false, af[mf], false, bf[nf], (short)0, acc[mf][nf], false, false);

    if (hasNext) store_tile(cur ^ 1);    // write other buffer: no 2nd barrier
    cur ^= 1;
  }

  // ---- epilogue: bias + activation + guarded store ----
  for (int mf = 0; mf < 4; ++mf) {
    for (int nf = 0; nf < 2; ++nf) {
      int col = n0 + wn * 32 + nf * 16 + lr;
      if (col >= g.N) continue;
      for (int v = 0; v < 8; ++v) {
        int row = m0 + wm * 64 + mf * 16 + lh * 8 + v;
        if (row >= g.M) continue;
        float x = acc[mf][nf][v];
        if (g.bias) x += g.bias[row];
        x = act_f(x, g.act);
        g.C[(size_t)row * g.N + col] = x;
      }
    }
  }
}

// ---------------------------------------------------------------------------
// Elementwise / layout kernels  (activation layout: [C][NBATCH][H][W])
// ---------------------------------------------------------------------------
__global__ void nchw_to_cnhw_kernel(const float* in, float* out,
                                    int C, int HW, long total) {
  long i = (long)blockIdx.x * 256 + threadIdx.x;
  if (i >= total) return;
  long c  = i / ((long)NBATCH * HW);
  long r  = i - c * (long)NBATCH * HW;
  long n  = r / HW;
  long hw = r - n * HW;
  out[i] = in[n * (long)C * HW + c * HW + hw];
}

__global__ void bn_kernel(float* x, const float* g, const float* b,
                          const float* mu, const float* var,
                          long NHW, long total) {
  long i = (long)blockIdx.x * 256 + threadIdx.x;
  if (i >= total) return;
  int c = (int)(i / NHW);
  float inv = rsqrtf(var[c] + 1e-5f);
  x[i] = (x[i] - mu[c]) * (g[c] * inv) + b[c];
}

__global__ void pool_kernel(const float* in, float* out,
                            int H, int W, int Ho, int Wo, long total) {
  long i = (long)blockIdx.x * 256 + threadIdx.x;
  if (i >= total) return;
  long cn = i / ((long)Ho * Wo);
  long hw = i - cn * (long)Ho * Wo;
  int ho = (int)(hw / Wo), wo = (int)(hw - (long)ho * Wo);
  const float* p = in + cn * (long)H * W + (long)(2 * ho) * W + 2 * wo;
  out[i] = fmaxf(fmaxf(p[0], p[1]), fmaxf(p[W], p[W + 1]));
}

__global__ void deconv_scatter_kernel(const float* t, float* out,
                                      const float* bias, int h, int w,
                                      int i, int j, long total) {
  long idx = (long)blockIdx.x * 256 + threadIdx.x;
  if (idx >= total) return;
  long cn = idx / ((long)h * w);              // fused [Co][NBATCH]
  long hw = idx - cn * (long)h * w;
  int hh = (int)(hw / w), ww = (int)(hw - (long)hh * w);
  int co = (int)(cn / NBATCH);
  float x = t[idx] + bias[co];
  out[(cn * (2L * h) + (2 * hh + i)) * (2L * w) + (2 * ww + j)] = selu_f(x);
}

__global__ void reduce_max_kernel(const float* v, int n, float* out) {
  __shared__ float s[256];
  float m = -1e30f;
  for (int i = threadIdx.x; i < n; i += 256) m = fmaxf(m, v[i]);
  s[threadIdx.x] = m; __syncthreads();
  for (int k = 128; k > 0; k >>= 1) {
    if ((int)threadIdx.x < k) s[threadIdx.x] = fmaxf(s[threadIdx.x], s[threadIdx.x + k]);
    __syncthreads();
  }
  if (threadIdx.x == 0) out[0] = s[0];
}

// ---------------------------------------------------------------------------
// Quantile mapping — stage 1: per-pixel sorts + CDFs + DNN input assembly
// ---------------------------------------------------------------------------
__device__ __forceinline__ int ubound256(const float* a, float x) {
  int lo = 0, hi = 256;
  while (lo < hi) { int mid = (lo + hi) >> 1; if (a[mid] <= x) lo = mid + 1; else hi = mid; }
  return lo;
}

__global__ __launch_bounds__(256)
void qm_prep_kernel(const float* last, const float* bobs, const float* all_max,
                    const float* max_value, const float* plat, const float* plon,
                    const float* u_mod, const float* u_obs, const float* maxv_p,
                    float* mod_s_ws, int* idx_ws, float* cdfmod_ws,
                    float* obsmax_ws, float* inp_ws) {
  const int p = blockIdx.x;
  const int n = threadIdx.x;
  __shared__ float s_mod[256]; __shared__ int s_idx[256];
  __shared__ float s_obs[256]; __shared__ float s_red[256];

  float gm   = all_max[p];
  float wide = gm / (float)CDFN;
  float lastv = last[(size_t)n * PPIX + p] * SCALE_C;
  float mod_raw = (lastv != lastv) ? 0.f : lastv;
  float bv  = bobs[(size_t)n * PPIX + p];
  float obs = (bv != bv) ? 0.f : bv;
  float um = u_mod[(size_t)n * PPIX + p];
  float uo = u_obs[(size_t)n * PPIX + p];
  float mod  = (mod_raw == 0.f) ? um * wide * 0.1f : mod_raw;
  float obsr = (obs     == 0.f) ? uo * wide * 0.1f : obs;

  s_red[n] = obs; __syncthreads();
  for (int k = 128; k > 0; k >>= 1) {
    if (n < k) s_red[n] = fmaxf(s_red[n], s_red[n + k]);
    __syncthreads();
  }
  if (n == 0) obsmax_ws[p] = s_red[0];

  s_mod[n] = mod; s_idx[n] = n; s_obs[n] = obsr;
  __syncthreads();

  for (int k = 2; k <= 256; k <<= 1) {
    for (int j = k >> 1; j > 0; j >>= 1) {
      int ixj = n ^ j;
      if (ixj > n) {
        bool up = ((n & k) == 0);
        float a0 = s_mod[n], a1 = s_mod[ixj];
        if (up ? (a0 > a1) : (a0 < a1)) {
          s_mod[n] = a1; s_mod[ixj] = a0;
          int t = s_idx[n]; s_idx[n] = s_idx[ixj]; s_idx[ixj] = t;
        }
        float b0 = s_obs[n], b1 = s_obs[ixj];
        if (up ? (b0 > b1) : (b0 < b1)) { s_obs[n] = b1; s_obs[ixj] = b0; }
      }
      __syncthreads();
    }
  }

  mod_s_ws[(size_t)p * 256 + n] = s_mod[n];
  idx_ws  [(size_t)p * 256 + n] = s_idx[n];
  __syncthreads();

  const float inv_n = 1.f / (float)NBATCH;
  int c0m = ubound256(s_mod, 0.f);
  int c0o = ubound256(s_obs, 0.f);
  for (int t = n; t <= CDFN; t += 256) {
    float cm, co;
    if (t == 0) { cm = 0.f; co = 0.f; }
    else {
      float xb = (float)t * wide;
      cm = (float)(ubound256(s_mod, xb) - c0m) * inv_n;
      co = (float)(ubound256(s_obs, xb) - c0o) * inv_n;
    }
    cdfmod_ws[(size_t)p * (CDFN + 1) + t] = cm;
    inp_ws[(size_t)t * PPIX + p] = co;
  }
  if (n == 0) {
    int rr = p / 48, cc = p - rr * 48;
    inp_ws[(size_t)(CDFN + 1) * PPIX + p] = plat[rr] / 48.534f;
    inp_ws[(size_t)(CDFN + 2) * PPIX + p] = plon[cc] / 293.75f;
    inp_ws[(size_t)(CDFN + 3) * PPIX + p] = max_value[p] / maxv_p[0];
  }
}

// ---------------------------------------------------------------------------
// Quantile mapping — stage 2: interp chain + scatter + output
// ---------------------------------------------------------------------------
__global__ __launch_bounds__(256)
void qm_final_kernel(const float* last, const float* all_max,
                     const float* mod_s_ws, const int* idx_ws,
                     const float* cdfmod_ws, const float* cobs_ws,
                     const float* obsmax_ws, float* xqm) {
  const int p = blockIdx.x;
  const int n = threadIdx.x;
  __shared__ float s_cm[CDFN + 1];
  __shared__ float s_co[CDFN + 1];
  __shared__ float s_a[256];
  for (int k = n; k <= CDFN; k += 256) {
    s_cm[k] = cdfmod_ws[(size_t)p * (CDFN + 1) + k];
    s_co[k] = cobs_ws[(size_t)k * PPIX + p];
  }
  __syncthreads();

  float gm = all_max[p], wide = gm / (float)CDFN;
  float v  = mod_s_ws[(size_t)p * 256 + n];
  int   id = idx_ws  [(size_t)p * 256 + n];

  int i1 = (int)ceilf(v / wide);
  i1 = i1 < 1 ? 1 : (i1 > CDFN ? CDFN : i1);
  float w1 = v / wide - (float)(i1 - 1);
  float cdf1 = (1.f - w1) * s_cm[i1 - 1] + w1 * s_cm[i1];

  int lo = 0, hi = CDFN + 1;
  while (lo < hi) { int mid = (lo + hi) >> 1; if (s_co[mid] < cdf1) lo = mid + 1; else hi = mid; }
  int i2 = lo < 1 ? 1 : (lo > CDFN ? CDFN : lo);
  float w2 = (cdf1 - s_co[i2 - 1]) / (s_co[i2] - s_co[i2 - 1]);
  float pre = (1.f - w2) * ((float)(i2 - 1) * wide) + w2 * ((float)i2 * wide);

  s_a[id] = pre;
  __syncthreads();
  float a = s_a[n];
  if (a != a) a = 0.f;
  float lastv = last[(size_t)n * PPIX + p] * SCALE_C;
  float mod_raw = (lastv != lastv) ? 0.f : lastv;
  xqm[(size_t)n * PPIX + p] = (obsmax_ws[p] > 0.f) ? a : 1e-12f * mod_raw;
}

// ---------------------------------------------------------------------------
// Host side
// ---------------------------------------------------------------------------
static void gemm(hipStream_t s, const float* A, int aBase, int sAm, int sAk,
                 const float* B, const float* bias, float* C,
                 int M, int K, int N, int act,
                 int conv = 0, int Hin = 1, int Win = 1, int kh = 1, int kw = 1,
                 int Ho = 1, int Wo = 1) {
  GemmArgs g{A, B, bias, C, M, K, N, sAm, sAk, aBase, act,
             Hin, Win, kh, kw, Ho, Wo, Hin * Win, NBATCH * Hin * Win, Ho * Wo};
  dim3 grid((N + 127) / 128, (M + 127) / 128);
  const bool avec = (sAk == 1) && ((sAm & 3) == 0) && ((aBase & 3) == 0);
  if (conv) {
    if (avec) gemm_wmma_kernel<1, 1><<<grid, dim3(256), 0, s>>>(g);
    else      gemm_wmma_kernel<1, 0><<<grid, dim3(256), 0, s>>>(g);
  } else {
    if (avec) gemm_wmma_kernel<0, 1><<<grid, dim3(256), 0, s>>>(g);
    else      gemm_wmma_kernel<0, 0><<<grid, dim3(256), 0, s>>>(g);
  }
}

static inline dim3 g1(long total) { return dim3((unsigned)((total + 255) / 256)); }

extern "C" void kernel_launch(void* const* d_in, const int* in_sizes, int n_in,
                              void* d_out, int out_size, void* d_ws, size_t ws_size,
                              hipStream_t stream) {
  (void)in_sizes; (void)n_in; (void)out_size; (void)ws_size;
  const float* x        = (const float*)d_in[0];
  const float* bobs     = (const float*)d_in[1];
  /* d_in[2] = cdfn == 500, fixed */
  const float* all_max  = (const float*)d_in[3];
  const float* max_val  = (const float*)d_in[4];
  const float* plat     = (const float*)d_in[5];
  const float* plon     = (const float*)d_in[6];
  const float* u_mod    = (const float*)d_in[7];
  const float* u_obs    = (const float*)d_in[8];
  const int PB = 9;
#define PW(i) ((const float*)d_in[PB + (i)])
  // 0 c1_w 1 c1_b 2 c11_w 3 c11_b 4 c2_w 5 c2_b 6 c22_w 7 c22_b
  // 8 c3_w 9 c3_b 10 c33_w 11 c33_b 12 c4_w 13 c4_b 14 c44_w 15 c44_b
  // 16 c5_w 17 c5_b 18 c6_w 19 c6_b 20 c66_w 21 c66_b 22 c7_w 23 c7_b
  // 24 c77_w 25 c77_b 26 c8_w 27 c8_b 28 c88_w 29 c88_b 30 c9_w 31 c9_b
  // 32 c99_w 33 c99_b 34 c10_w 35 c10_b 36 m_w 37 m_b 38 m1_w 39 m1_b
  // 40 b1_w 41 b1_b 42 b2_w 43 b2_b 44 b3_w 45 b3_b 46 b4_w 47 b4_b
  // 48 b5_w 49 b5_b 50 b6_w 51 b6_b 52 b7_w 53 b7_b
  // 54 d1_w 55 d1_b 56 d2_w 57 d2_b 58 d3_w 59 d3_b 60 d4_w 61 d4_b
  // 62-65 bn8 g/b/m/v  66-69 bn9  70-73 bn10
  // 74 dnn_w1 75 dnn_b1 76 dnn_w2 77 dnn_b2

  const long NHW0 = (long)NBATCH * 36 * 56;
  const long NHW1 = (long)NBATCH * 18 * 28;
  const long NHW2 = (long)NBATCH * 9 * 14;
  const long NHW3 = (long)NBATCH * 4 * 7;
  const long NHW4 = (long)NBATCH * 2 * 3;

  float* ws = (float*)d_ws;
  size_t off = 0;
  auto alloc = [&](size_t nfl) { float* p = ws + off; off += nfl; return p; };

  float* xc    = alloc(6 * NHW0);
  float* x0    = alloc((size_t)120 * NHW0);
  float* x1    = alloc((size_t)120 * NHW1);
  float* x2    = alloc((size_t)120 * NBATCH * 32 * 48);  // cat head
  float* u4    = alloc((size_t)120 * NBATCH * 32 * 48);  // [x2;u4] contiguous
  float* x11   = alloc((size_t)240 * NHW1);
  float* x12   = alloc((size_t)240 * NHW2);
  float* x3    = alloc((size_t)240 * NBATCH * 16 * 24);
  float* u3    = alloc((size_t)240 * NBATCH * 16 * 24);  // [x3;u3]
  float* x121  = alloc((size_t)480 * NHW2);
  float* x122  = alloc((size_t)480 * NHW3);
  float* x4    = alloc((size_t)480 * NBATCH * 8 * 12);
  float* u2    = alloc((size_t)480 * NBATCH * 8 * 12);   // [x4;u2]
  float* x1221 = alloc((size_t)960 * NHW3);
  float* x1222 = alloc((size_t)960 * NHW4);
  float* x5    = alloc((size_t)960 * NBATCH * 4 * 6);
  float* u1    = alloc((size_t)960 * NBATCH * 4 * 6);    // [x5;u1]
  float* x12221= alloc((size_t)960 * NHW4);
  float* x7    = alloc((size_t)960 * NBATCH * 4 * 6);
  float* x10   = alloc((size_t)480 * NBATCH * 8 * 12);
  float* x13   = alloc((size_t)240 * NBATCH * 16 * 24);
  float* x16   = alloc((size_t)120 * NBATCH * 32 * 48);
  float* t10   = alloc((size_t)10 * NBATCH * 32 * 48);
  float* tm    = alloc((size_t)10 * NBATCH * 28 * 48);
  float* mods  = alloc((size_t)PPIX * 256);
  int*   idxw  = (int*)alloc((size_t)PPIX * 256);
  float* cdfm  = alloc((size_t)PPIX * (CDFN + 1));
  float* obsm  = alloc(PPIX);
  float* inpw  = alloc((size_t)(CDFN + 4) * PPIX);
  float* cobs  = alloc((size_t)(CDFN + 1) * PPIX);
  float* maxv  = alloc(16);
  float* tmp   = alloc((size_t)120 * NHW0);  // shared temp (largest user: c1)

  float* outp  = (float*)d_out;              // [0..319487]=last, rest=x_qm

  // ---------------- U-Net ----------------
  nchw_to_cnhw_kernel<<<g1(6 * NHW0), 256, 0, stream>>>(x, xc, 6, 36 * 56, 6 * NHW0);

  gemm(stream, PW(0), 0, 6, 1, xc, PW(1), tmp, 120, 6, (int)NHW0, 2);       // c1+selu
  gemm(stream, PW(2), 0, 120, 1, tmp, PW(3), x0, 120, 120, (int)NHW0, 2);   // c11+selu
  bn_kernel<<<g1(120 * NHW0), 256, 0, stream>>>(x0, PW(62), PW(63), PW(64), PW(65),
                                                NHW0, 120 * NHW0);          // bn8
  pool_kernel<<<g1(120 * NHW1), 256, 0, stream>>>(x0, x1, 36, 56, 18, 28, 120 * NHW1);

  gemm(stream, PW(40), 0, 120 * 18, 1, x0, PW(41), tmp, 120, 120 * 18,
       NBATCH * 34 * 51, 0, 1, 36, 56, 3, 6, 34, 51);                       // b1
  gemm(stream, PW(42), 0, 120 * 12, 1, tmp, PW(43), x2, 120, 120 * 12,
       NBATCH * 32 * 48, 2, 1, 34, 51, 3, 4, 32, 48);                       // b2+selu

  gemm(stream, PW(4), 0, 120, 1, x1, PW(5), tmp, 240, 120, (int)NHW1, 2);   // c2
  gemm(stream, PW(6), 0, 240, 1, tmp, PW(7), x11, 240, 240, (int)NHW1, 2);  // c22
  pool_kernel<<<g1(240 * NHW2), 256, 0, stream>>>(x11, x12, 18, 28, 9, 14, 240 * NHW2);

  gemm(stream, PW(44), 0, 240 * 8, 1, x11, PW(45), tmp, 240, 240 * 8,
       NBATCH * 17 * 25, 0, 1, 18, 28, 2, 4, 17, 25);                       // b3
  gemm(stream, PW(46), 0, 240 * 4, 1, tmp, PW(47), x3, 240, 240 * 4,
       NBATCH * 16 * 24, 2, 1, 17, 25, 2, 2, 16, 24);                       // b4+selu

  gemm(stream, PW(8), 0, 240, 1, x12, PW(9), tmp, 480, 240, (int)NHW2, 2);  // c3
  gemm(stream, PW(10), 0, 480, 1, tmp, PW(11), x121, 480, 480, (int)NHW2, 2);// c33
  bn_kernel<<<g1(480 * NHW2), 256, 0, stream>>>(x121, PW(66), PW(67), PW(68), PW(69),
                                                NHW2, 480 * NHW2);          // bn9
  pool_kernel<<<g1(480 * NHW3), 256, 0, stream>>>(x121, x122, 9, 14, 4, 7, 480 * NHW3);

  gemm(stream, PW(48), 0, 480 * 2, 1, x121, PW(49), tmp, 480, 480 * 2,
       NBATCH * 9 * 13, 0, 1, 9, 14, 1, 2, 9, 13);                          // b5
  gemm(stream, PW(50), 0, 480 * 4, 1, tmp, PW(51), x4, 480, 480 * 4,
       NBATCH * 8 * 12, 2, 1, 9, 13, 2, 2, 8, 12);                          // b6+selu

  gemm(stream, PW(12), 0, 480, 1, x122, PW(13), tmp, 960, 480, (int)NHW3, 2);// c4
  gemm(stream, PW(14), 0, 960, 1, tmp, PW(15), x1221, 960, 960, (int)NHW3, 2);// c44
  bn_kernel<<<g1(960 * NHW3), 256, 0, stream>>>(x1221, PW(70), PW(71), PW(72), PW(73),
                                                NHW3, 960 * NHW3);          // bn10
  pool_kernel<<<g1(960 * NHW4), 256, 0, stream>>>(x1221, x1222, 4, 7, 2, 3, 960 * NHW4);

  gemm(stream, PW(52), 0, 960 * 2, 1, x1221, PW(53), x5, 960, 960 * 2,
       NBATCH * 4 * 6, 2, 1, 4, 7, 1, 2, 4, 6);                             // b7+selu

  gemm(stream, PW(16), 0, 960, 1, x1222, PW(17), tmp, 960, 960, (int)NHW4, 2);// c5
  gemm(stream, PW(16), 0, 960, 1, tmp, PW(17), x12221, 960, 960, (int)NHW4, 2);// c5 again

  for (int ij = 0; ij < 4; ++ij) {
    gemm(stream, PW(54), ij, 4, 960 * 4, x12221, nullptr, tmp, 960, 960, (int)NHW4, 0);
    deconv_scatter_kernel<<<g1(960 * NHW4), 256, 0, stream>>>(
        tmp, u1, PW(55), 2, 3, ij >> 1, ij & 1, 960 * NHW4);
  }
  gemm(stream, PW(18), 0, 1920, 1, x5, PW(19), tmp, 960, 1920, NBATCH * 4 * 6, 1); // c6 (cat)
  gemm(stream, PW(20), 0, 960, 1, tmp, PW(21), x7, 960, 960, NBATCH * 4 * 6, 1);   // c66

  for (int ij = 0; ij < 4; ++ij) {
    gemm(stream, PW(56), ij, 4, 480 * 4, x7, nullptr, tmp, 480, 960, NBATCH * 4 * 6, 0);
    deconv_scatter_kernel<<<g1((long)480 * NBATCH * 4 * 6), 256, 0, stream>>>(
        tmp, u2, PW(57), 4, 6, ij >> 1, ij & 1, (long)480 * NBATCH * 4 * 6);
  }
  gemm(stream, PW(22), 0, 960, 1, x4, PW(23), tmp, 480, 960, NBATCH * 8 * 12, 1);  // c7 (cat)
  gemm(stream, PW(24), 0, 480, 1, tmp, PW(25), x10, 480, 480, NBATCH * 8 * 12, 1); // c77

  for (int ij = 0; ij < 4; ++ij) {
    gemm(stream, PW(58), ij, 4, 240 * 4, x10, nullptr, tmp, 240, 480, NBATCH * 8 * 12, 0);
    deconv_scatter_kernel<<<g1((long)240 * NBATCH * 8 * 12), 256, 0, stream>>>(
        tmp, u3, PW(59), 8, 12, ij >> 1, ij & 1, (long)240 * NBATCH * 8 * 12);
  }
  gemm(stream, PW(26), 0, 480, 1, x3, PW(27), tmp, 240, 480, NBATCH * 16 * 24, 1); // c8 (cat)
  gemm(stream, PW(28), 0, 240, 1, tmp, PW(29), x13, 240, 240, NBATCH * 16 * 24, 1);// c88

  for (int ij = 0; ij < 4; ++ij) {
    gemm(stream, PW(60), ij, 4, 120 * 4, x13, nullptr, tmp, 120, 240, NBATCH * 16 * 24, 0);
    deconv_scatter_kernel<<<g1((long)120 * NBATCH * 16 * 24), 256, 0, stream>>>(
        tmp, u4, PW(61), 16, 24, ij >> 1, ij & 1, (long)120 * NBATCH * 16 * 24);
  }
  gemm(stream, PW(30), 0, 240, 1, x2, PW(31), tmp, 120, 240, NBATCH * 32 * 48, 1); // c9 (cat)
  gemm(stream, PW(32), 0, 120, 1, tmp, PW(33), x16, 120, 120, NBATCH * 32 * 48, 1);// c99

  gemm(stream, PW(34), 0, 120, 1, x16, PW(35), t10, 10, 120, NBATCH * 32 * 48, 0); // c10
  gemm(stream, PW(36), 0, 50, 1, t10, PW(37), tm, 10, 50, NBATCH * 28 * 48,
       2, 1, 32, 48, 5, 1, 28, 48);                                         // m+selu
  gemm(stream, PW(38), 0, 30, 1, tm, PW(39), outp, 1, 30, NBATCH * 26 * 48,
       1, 1, 28, 48, 3, 1, 26, 48);                                         // m1+relu -> last

  // ---------------- quantile mapping ----------------
  reduce_max_kernel<<<1, 256, 0, stream>>>(max_val, PPIX, maxv);
  qm_prep_kernel<<<PPIX, 256, 0, stream>>>(outp, bobs, all_max, max_val, plat, plon,
                                           u_mod, u_obs, maxv,
                                           mods, idxw, cdfm, obsm, inpw);
  gemm(stream, PW(74), 0, 504, 1, inpw, PW(75), tmp, 512, 504, PPIX, 1);    // dnn1+relu
  gemm(stream, PW(76), 0, 512, 1, tmp, PW(77), cobs, 501, 512, PPIX, 3);    // dnn2+sigmoid
  qm_final_kernel<<<PPIX, 256, 0, stream>>>(outp, all_max, mods, idxw, cdfm, cobs,
                                            obsm, outp + (size_t)NBATCH * PPIX);
#undef PW
}
